// NaryLayer_4458176053338
// MI455X (gfx1250) — matrix operand order repaired and checked by reading, not compiled, not run on hardware
//
#include <hip/hip_runtime.h>
#include <hip/hip_bf16.h>

// ---------------------------------------------------------------------------
// Tree-LSTM (N-ary, K=4 children) forward for MI455X (gfx1250, wave32).
// GEMMs: v_wmma_f32_16x16x32_bf16, fed by a 2-deep async pipeline:
//   A tiles (gathered child rows): global_load_async_to_lds_b128 (ASYNCcnt)
//   B tiles (weights, uniform 2D): TDM tensor_load_to_lds        (TENSORcnt)
// ---------------------------------------------------------------------------

typedef __attribute__((ext_vector_type(16))) __bf16   v16bf;
typedef __attribute__((ext_vector_type(8)))  float    v8f;
typedef __attribute__((ext_vector_type(8)))  unsigned v8u;
typedef __attribute__((ext_vector_type(4)))  int      v4i;
typedef __attribute__((ext_vector_type(8)))  int      v8i;

#define DD      12
#define NNODES  32768
#define LTOK    3
#define KCH     4
#define HID     128
#define VOC     50000
#define NLAYERS 2

union FragBF {
  v16bf bf;
  v8u   u;
};

static __device__ __forceinline__ __bf16 f2bf(float x) {
  union { float f; unsigned u; } v; v.f = x;
  unsigned r = v.u + 0x7FFFu + ((v.u >> 16) & 1u);   // round-to-nearest-even
  unsigned short h = (unsigned short)(r >> 16);
  __bf16 out;
  __builtin_memcpy(&out, &h, sizeof(out));
  return out;
}

// ---------------------------------------------------------------------------
// Elementwise helpers
// ---------------------------------------------------------------------------
__global__ void cvt_f32_bf16(const float* __restrict__ src, __bf16* __restrict__ dst, long n) {
  long i = (long)blockIdx.x * blockDim.x + threadIdx.x;
  if (i < n) dst[i] = f2bf(src[i]);
}

__global__ void zero_bf16_k(__bf16* __restrict__ dst, long n) {
  long i = (long)blockIdx.x * blockDim.x + threadIdx.x;
  if (i < n) dst[i] = f2bf(0.0f);
}

__global__ void zero_f32_k(float* __restrict__ dst, long n) {
  long i = (long)blockIdx.x * blockDim.x + threadIdx.x;
  if (i < n) dst[i] = 0.0f;
}

// ---------------------------------------------------------------------------
// WMMA fragment loaders (LDS row stride = 136 bf16: 256B data + 16B pad).
// A-matrix 16x32 bf16 layout (ISA 7.12.2): lane m = lane&15, half = lane>>4,
//   VGPR r (r<4): K = 8*half + 2r ; (r>=4): K = 16 + 8*half + 2(r-4)
// B-matrix 32x16 bf16: lane n = lane&15, half = lane>>4, VGPR r: same K as A
// ---------------------------------------------------------------------------
#define LDS_STRIDE 136

static __device__ __forceinline__ v16bf load_frag_a(const __bf16* s, int m0, int k0, int lane) {
  int m = lane & 15, half = lane >> 4;
  const unsigned* row = (const unsigned*)(s + (long)(m0 + m) * LDS_STRIDE);
  FragBF f;
#pragma unroll
  for (int r = 0; r < 8; ++r) {
    int k = (r < 4) ? (8 * half + 2 * r) : (16 + 8 * half + 2 * (r - 4));
    f.u[r] = row[(k0 + k) >> 1];
  }
  return f.bf;
}

static __device__ __forceinline__ v16bf load_frag_b(const __bf16* s, int n0, int k0, int lane) {
  int n = lane & 15, half = lane >> 4;
  const unsigned* row = (const unsigned*)(s + (long)(n0 + n) * LDS_STRIDE);
  FragBF f;
#pragma unroll
  for (int r = 0; r < 8; ++r) {
    int k = (r < 4) ? (8 * half + 2 * r) : (16 + 8 * half + 2 * (r - 4));
    f.u[r] = row[(k0 + k) >> 1];
  }
  return f.bf;
}

// ---------------------------------------------------------------------------
// Generic gather-A WMMA GEMM:  C[M][Ncols] = gatherA @ W^T + bias
//   A rows: 128-wide bf16 rows in Atab; K-chunk c uses row
//     max(rowidx[m*nchunks + c], 0)   (or m itself when rowidx==null, nchunks==1)
//   W: [Ncols][K] row-major bf16, K = 128*nchunks.
// Block tile 64(M) x 64(N); 8 waves: 4 along M x 2 along N, 16x32 per wave.
// Double-buffered LDS; A via async-to-LDS, B via TDM descriptor (wave 0).
// ---------------------------------------------------------------------------
__global__ __launch_bounds__(256)
void gemm_bf16_wmma(const __bf16* __restrict__ Atab,
                    const int* __restrict__ rowidx, int nchunks,
                    const __bf16* __restrict__ W, const float* __restrict__ bias,
                    float* __restrict__ C, int M, int Ncols) {
  __shared__ __align__(16) __bf16 sA[2][64 * LDS_STRIDE];
  __shared__ __align__(16) __bf16 sB[2][64 * LDS_STRIDE];
  const int mblk = blockIdx.x * 64;
  const int nblk = blockIdx.y * 64;
  const int tid  = threadIdx.x;
  const int lane = tid & 31;
  const int wid  = tid >> 5;
  const int wr   = wid >> 1;  // 0..3 : wave row (M)
  const int wc   = wid & 1;   // 0..1 : wave col (N)
  const long Kl  = (long)nchunks * 128;

  v8f acc0 = {};
  v8f acc1 = {};

  const int ldr = tid >> 4;         // 0..15 : staging row
  const int ldc = (tid & 15) * 8;   // staging col (elements), 16B granularity

  // --- TDM D# group 1 (uniform, constant over K loop) -----------------------
  // [17:16] data_size=1 (2B)  [20] pad_enable  [24:22] pad_interval=5 (64 dw)
  // [31:25] pad_amount=3 (4 dw = 16B)  [79:48] tensor_dim0  [111:80] tensor_dim1
  // [127:112] tile_dim0=128  [143:128] tile_dim1=64  [207:160] dim0_stride=K
  const unsigned tdim0 = 0x40000000u, tdim1 = 0x40000000u;
  v8i g1;
  g1[0] = (int)((1u << 16) | (1u << 20) | (5u << 22) | (3u << 25));
  g1[1] = (int)((tdim0 & 0xffffu) << 16);
  g1[2] = (int)((tdim0 >> 16) | ((tdim1 & 0xffffu) << 16));
  g1[3] = (int)((tdim1 >> 16) | (128u << 16));
  g1[4] = 64;                                   // tile_dim1=64, tile_dim2=0
  g1[5] = (int)(unsigned)(Kl & 0xffffffff);     // dim0_stride low 32
  g1[6] = (int)(unsigned)((Kl >> 32) & 0xffff); // dim0_stride high 16, stride1 lo=0
  g1[7] = 0;

  auto issue_tiles = [&](int chunk, int buf) {
    // ---- A tile: per-lane async global->LDS (4 x b128 per thread group) ----
#pragma unroll
    for (int it = 0; it < 4; ++it) {
      const int row = ldr + it * 16;            // 0..63
      const int gm  = mblk + row;
      long grow;
      if (rowidx) {
        int rv = rowidx[(long)gm * nchunks + chunk];
        grow = (long)(rv < 0 ? 0 : rv);
      } else {
        grow = (long)gm;
      }
      unsigned long long ga =
          (unsigned long long)(size_t)(Atab + grow * 128 + ldc);
      unsigned la = (unsigned)(size_t)(&sA[buf][0]) +
                    (unsigned)((row * LDS_STRIDE + ldc) * 2);
      asm volatile("global_load_async_to_lds_b128 %0, %1, off"
                   :: "v"(la), "v"(ga) : "memory");
    }
    // ---- B tile: one TDM descriptor, issued by wave 0 ----------------------
    if (wid == 0) {
      unsigned long long ga =
          (unsigned long long)(size_t)(W + (size_t)nblk * Kl + (size_t)chunk * 128);
      v4i g0;
      g0[0] = 1;                                            // count=1
      g0[1] = (int)(unsigned)(size_t)(&sB[buf][0]);         // lds_addr
      g0[2] = (int)(unsigned)(ga & 0xffffffffu);            // global_addr[31:0]
      g0[3] = (int)((((unsigned)(ga >> 32)) & 0x1ffffffu)   // global_addr[56:32]
                    | 0x80000000u);                         // type=2
      asm volatile("tensor_load_to_lds %0, %1" :: "s"(g0), "s"(g1) : "memory");
    }
  };

  issue_tiles(0, 0);  // prologue

  for (int chunk = 0; chunk < nchunks; ++chunk) {
    const int buf = chunk & 1;
    const bool more = (chunk + 1 < nchunks);
    if (more) issue_tiles(chunk + 1, buf ^ 1);

    // wait for THIS chunk's tiles; leave next chunk's transfers in flight
    if (more) {
      asm volatile("s_wait_asynccnt 0x4" ::: "memory");
      if (wid == 0) __builtin_amdgcn_s_wait_tensorcnt(1);
    } else {
      asm volatile("s_wait_asynccnt 0x0" ::: "memory");
      if (wid == 0) __builtin_amdgcn_s_wait_tensorcnt(0);
    }
    __syncthreads();

#pragma unroll
    for (int kk = 0; kk < 4; ++kk) {
      v16bf a  = load_frag_a(sA[buf], wr * 16, kk * 32, lane);
      v16bf b0 = load_frag_b(sB[buf], wc * 32, kk * 32, lane);
      v16bf b1 = load_frag_b(sB[buf], wc * 32 + 16, kk * 32, lane);
      acc0 = __builtin_amdgcn_wmma_f32_16x16x32_bf16(false, a, false, b0, (short)0, acc0, false, false);
      acc1 = __builtin_amdgcn_wmma_f32_16x16x32_bf16(false, a, false, b1, (short)0, acc1, false, false);
    }
    __syncthreads();  // all reads done before next chunk overwrites this buf
  }

  // C/D layout: lanes 0-15 -> N=lane, M=r ; lanes 16-31 -> N=lane-16, M=8+r
  const int nloc = lane & 15;
  const int half = lane >> 4;
#pragma unroll
  for (int r = 0; r < 8; ++r) {
    long m  = (long)mblk + wr * 16 + half * 8 + r;
    int  c0 = nblk + wc * 32 + nloc;
    int  c1 = c0 + 16;
    C[m * Ncols + c0] = acc0[r] + bias[c0];
    C[m * Ncols + c1] = acc1[r] + bias[c1];
  }
}

// ---------------------------------------------------------------------------
// Gate math per node-element. Exact N*HID threads.
// ---------------------------------------------------------------------------
__global__ __launch_bounds__(256)
void gate_kernel(const float* __restrict__ wx, const float* __restrict__ uo,
                 const int* __restrict__ idx,
                 const float* __restrict__ cin, float* __restrict__ cout,
                 __bf16* __restrict__ htab, float* __restrict__ td) {
  const int g = blockIdx.x * blockDim.x + threadIdx.x;
  const int n = g >> 7;
  const int e = g & 127;
  const float* wrow = wx + (long)n * 512;
  const float* urow = uo + (long)n * 896;
  const float Wf = wrow[e], Wi = wrow[128 + e], Wu = wrow[256 + e], Wo = wrow[384 + e];

  float f = 0.0f;
#pragma unroll
  for (int k = 0; k < KCH; ++k) {
    int j = idx[(long)n * KCH + k];
    int safe = j < 0 ? 0 : j;                  // row 0 is zeros -> mask implicit
    float fk = 1.0f / (1.0f + __expf(-(Wf + urow[k * 128 + e])));
    f += fk * cin[(long)safe * 128 + e];
  }
  const float i = 1.0f / (1.0f + __expf(-(urow[512 + e] + Wi)));
  const float u = tanhf(urow[640 + e] + Wu);
  const float o = 1.0f / (1.0f + __expf(-(urow[768 + e] + Wo)));
  const float nc = i * u + f;
  const float nh = o * tanhf(nc);

  cout[(long)(n + 1) * 128 + e] = nc;
  htab[(long)(n + 1) * 128 + e] = f2bf(nh);
  td[g] += nh;
}

// out: [hx(2,N,128) | cx(2,N,128)] ; ct rows 1..N hold final c
__global__ void out_copy(const float* __restrict__ tl, const float* __restrict__ ct,
                         float* __restrict__ o, long nh) {
  long g = (long)blockIdx.x * blockDim.x + threadIdx.x;
  if (g >= nh) return;
  float h = tl[g];
  float c = ct[128 + g];
  o[g]          = h;
  o[nh + g]     = h;
  o[2 * nh + g] = c;
  o[3 * nh + g] = c;
}

// ---------------------------------------------------------------------------
// Host driver
// ---------------------------------------------------------------------------
static inline long cdiv(long a, long b) { return (a + b - 1) / b; }

extern "C" void kernel_launch(void* const* d_in, const int* in_sizes, int n_in,
                              void* d_out, int out_size, void* d_ws, size_t ws_size,
                              hipStream_t stream) {
  (void)in_sizes; (void)n_in; (void)out_size; (void)ws_size;

  const int*   tokens  = (const int*)d_in[0];    // (D,N,L)
  const int*   indices = (const int*)d_in[1];    // (D,N,K)
  const float* E       = (const float*)d_in[2];  // (V,HID)
  const float* lin_w   = (const float*)d_in[3];  // (HID, L*HID)
  const float* lin_b   = (const float*)d_in[4];  // (HID,)
  const float* Uf_w    = (const float*)d_in[5];  // (2, 512, 512)
  const float* Uf_b    = (const float*)d_in[6];  // (2, 512)
  const float* Uiuo_w  = (const float*)d_in[7];  // (2, 384, 512)
  const float* Uiuo_b  = (const float*)d_in[8];  // (2, 384)
  const float* W_w     = (const float*)d_in[9];  // (2, 512, 128)
  const float* W_b     = (const float*)d_in[10]; // (2, 512)
  float* out = (float*)d_out;

  // ---- workspace carve-out (256B aligned slabs) ----
  char* ws = (char*)d_ws;
  size_t off = 0;
  auto alloc = [&](size_t bytes) -> void* {
    void* p = ws + off;
    off += bytes;
    off = (off + 255) & ~(size_t)255;
    return p;
  };
  __bf16* Ebf    = (__bf16*)alloc((size_t)VOC * HID * 2);
  __bf16* linwbf = (__bf16*)alloc((size_t)HID * (LTOK * HID) * 2);
  __bf16* Wbf    = (__bf16*)alloc((size_t)NLAYERS * 4 * HID * HID * 2);
  __bf16* Ucat   = (__bf16*)alloc((size_t)NLAYERS * 7 * HID * (KCH * HID) * 2);
  float*  bcat   = (float*) alloc((size_t)NLAYERS * 7 * HID * 4);
  __bf16* xbf    = (__bf16*)alloc((size_t)NNODES * HID * 2);
  __bf16* htab   = (__bf16*)alloc((size_t)(NNODES + 1) * HID * 2);
  float*  ctab0  = (float*) alloc((size_t)(NNODES + 1) * HID * 4);
  float*  ctab1  = (float*) alloc((size_t)(NNODES + 1) * HID * 4);
  float*  wx     = (float*) alloc((size_t)NNODES * 4 * HID * 4);
  float*  uout   = (float*) alloc((size_t)NNODES * 7 * HID * 4);
  float*  t      = (float*) alloc((size_t)DD * NNODES * HID * 4);
  float* ctabs[2] = {ctab0, ctab1};

  const int TB = 256;
#define GRD(n) dim3((unsigned)cdiv((long)(n), TB)), dim3(TB), 0, stream

  // ---- weight prep: f32 -> bf16 ----
  cvt_f32_bf16<<<GRD((long)VOC * HID)>>>(E, Ebf, (long)VOC * HID);
  cvt_f32_bf16<<<GRD((long)HID * LTOK * HID)>>>(lin_w, linwbf, (long)HID * LTOK * HID);
  cvt_f32_bf16<<<GRD((long)NLAYERS * 4 * HID * HID)>>>(W_w, Wbf, (long)NLAYERS * 4 * HID * HID);
  for (int l = 0; l < NLAYERS; ++l) {
    __bf16* ucl = Ucat + (size_t)l * 896 * 512;
    cvt_f32_bf16<<<GRD(512L * 512)>>>(Uf_w + (size_t)l * 512 * 512, ucl, 512L * 512);
    cvt_f32_bf16<<<GRD(384L * 512)>>>(Uiuo_w + (size_t)l * 384 * 512, ucl + (size_t)512 * 512, 384L * 512);
    hipMemcpyAsync(bcat + (size_t)l * 896, Uf_b + (size_t)l * 512, 512 * sizeof(float),
                   hipMemcpyDeviceToDevice, stream);
    hipMemcpyAsync(bcat + (size_t)l * 896 + 512, Uiuo_b + (size_t)l * 384, 384 * sizeof(float),
                   hipMemcpyDeviceToDevice, stream);
  }

  // ---- embedding GEMM: t[D*N][128] = gather(E, tokens)[*,384] @ lin_w^T + lin_b ----
  {
    dim3 g((unsigned)(DD * NNODES / 64), (unsigned)(HID / 64));
    gemm_bf16_wmma<<<g, 256, 0, stream>>>(Ebf, tokens, LTOK, linwbf, lin_b, t,
                                          DD * NNODES, HID);
  }

  // ---- layers ----
  const long NH = (long)NNODES * HID;
  for (int l = 0; l < NLAYERS; ++l) {
    zero_bf16_k<<<GRD((long)(NNODES + 1) * HID)>>>(htab, (long)(NNODES + 1) * HID);
    zero_f32_k<<<GRD((long)(NNODES + 1) * HID)>>>(ctab0, (long)(NNODES + 1) * HID);
    zero_f32_k<<<GRD((long)HID)>>>(ctab1, (long)HID);

    for (int d = 0; d < DD; ++d) {
      const int* idx_d = indices + (size_t)d * NNODES * KCH;
      float* td = t + (size_t)d * NH;

      // x = bf16(t[d]) (pre-residual value of this step)
      cvt_f32_bf16<<<GRD(NH)>>>(td, xbf, NH);

      // Wx = x @ W^T + W_b        (N x 128 x 512)
      {
        dim3 g((unsigned)(NNODES / 64), (unsigned)(512 / 64));
        gemm_bf16_wmma<<<g, 256, 0, stream>>>(xbf, nullptr, 1,
                                              Wbf + (size_t)l * 512 * HID,
                                              W_b + (size_t)l * 512, wx, NNODES, 512);
      }
      // Uout = gather(htab, idx) @ [Uf;Uiuo]^T + bias   (N x 512 x 896)
      {
        dim3 g((unsigned)(NNODES / 64), (unsigned)(896 / 64));
        gemm_bf16_wmma<<<g, 256, 0, stream>>>(htab, idx_d, KCH,
                                              Ucat + (size_t)l * 896 * 512,
                                              bcat + (size_t)l * 896, uout, NNODES, 896);
      }
      // gates + residual + carry tables (ping-pong c to avoid gather/write race)
      gate_kernel<<<GRD(NH)>>>(wx, uout, idx_d, ctabs[d & 1], ctabs[(d + 1) & 1],
                               htab, td);
    }
  }

  // final c parity: last write at d=11 -> parity (12)&1 = 0
  out_copy<<<GRD(NH)>>>(t + (size_t)(DD - 1) * NH, ctab0, out, NH);
#undef GRD
}
